// BaseSpatioTemporalHawkes_36782099923570
// MI455X (gfx1250) — matrix operand order repaired and checked by reading, not compiled, not run on hardware
//
#include <hip/hip_runtime.h>
#include <hip/hip_bf16.h>

typedef __attribute__((ext_vector_type(16))) _Float16 v16h;
typedef __attribute__((ext_vector_type(8)))  float    v8f;

#define BQ 8        // batches
#define LQ 64       // events per batch
#define NTI 50      // INT_RES time points
#define NGP 2500    // spatial grid points (50*50)
#define GTILES 157  // ceil(2500/16)
#define PI_F 3.14159265358979323846f
#define UNIT_VOL_F (1.0f / (50.0f * 50.0f * 50.0f))

__device__ __forceinline__ float softplus_f(float v) {
    // numerically stable softplus, matches jax.nn.softplus behavior
    return fmaxf(v, 0.0f) + log1pf(expf(-fabsf(v)));
}

// ---------------------------------------------------------------------------
// Kernel 1: lams[b,i] = softplus(mu + norm * sum_{j<i, t_j>0} exp(...)) + 1e-5
//           sumlog[b] = sum_i log(lams[b,i]) * (t_i > 0)
// Full-precision path (feeds f32 outputs directly). 8 blocks x 64 threads.
// ---------------------------------------------------------------------------
__global__ void k_lams(const float* __restrict__ x,
                       const float* __restrict__ pmu, const float* __restrict__ palpha,
                       const float* __restrict__ pbeta, const float* __restrict__ psigma,
                       float* __restrict__ lams_out, float* __restrict__ sumlog) {
    int b = blockIdx.x;
    int i = threadIdx.x;
    __shared__ float t[LQ], sx[LQ], sy[LQ];
    __shared__ float red[LQ];
    const float* xb = x + (size_t)b * LQ * 3;
    t[i]  = xb[i * 3 + 0];
    sx[i] = xb[i * 3 + 1];
    sy[i] = xb[i * 3 + 2];
    __syncthreads();

    float mu = *pmu, alpha = *palpha, beta = *pbeta, sigma = *psigma;
    float inv2s2 = 1.0f / (2.0f * sigma * sigma);
    float norm   = alpha * beta / (2.0f * PI_F * sigma * sigma);

    float ti = t[i], sxi = sx[i], syi = sy[i];
    float acc = 0.0f;
    for (int j = 0; j < i; ++j) {           // causal: strictly lower triangular
        if (t[j] > 0.0f) {                   // hist mask
            float dt = ti - t[j];
            float dx = sxi - sx[j];
            float dy = syi - sy[j];
            acc += expf(-beta * dt - (dx * dx + dy * dy) * inv2s2);
        }
    }
    float lam = softplus_f(norm * acc + mu) + 1e-5f;
    lams_out[b * LQ + i] = lam;

    red[i] = (ti > 0.0f) ? logf(lam) : 0.0f;
    __syncthreads();
    for (int s = 32; s > 0; s >>= 1) {
        if (i < s) red[i] += red[i + s];
        __syncthreads();
    }
    if (i == 0) sumlog[b] = red[0];
}

// ---------------------------------------------------------------------------
// Kernel 2: precompute temporal factor T[b, ti, j] = mask * exp(-beta*(tt-t_j))
// stored directly in the 16-bit A-matrix WMMA fragment layout (f16):
//   record index = ((b*4 + rowtile)*2 + frag)*32 + lane ; 16 halves per record
//   lane: m = lane&15 (row within tile), h = lane>>4
//   element e -> K = (e<8 ? 0 : 16) + h*8 + (e&7);  j = frag*32 + K
// Rows with ti >= 50 are stored as exact zeros (safe padding for WMMA).
// Result is rounded to f16, so native-rate __expf (v_exp_f32) loses nothing.
// 64 blocks x 32 threads
// ---------------------------------------------------------------------------
__global__ void k_tfrag(const float* __restrict__ x,
                        const float* __restrict__ pbeta,
                        _Float16* __restrict__ Tw) {
    int blk = blockIdx.x;             // ((b*4 + r)*2 + f)
    int l   = threadIdx.x;
    int f = blk & 1;
    int r = (blk >> 1) & 3;
    int b = blk >> 3;
    int h = l >> 4, m = l & 15;
    int tiIdx = r * 16 + m;
    float beta = *pbeta;
    float tt = (float)tiIdx * (1.0f / 49.0f);   // linspace(0,1,50)
    const float* xb = x + (size_t)b * LQ * 3;

    v16h av;
#pragma unroll
    for (int e = 0; e < 16; ++e) {
        int K = ((e < 8) ? 0 : 16) + h * 8 + (e & 7);
        int j = f * 32 + K;
        float tj = xb[j * 3];
        float v = 0.0f;
        if (tiIdx < NTI && tj > 0.0f && tj <= tt)
            v = __expf(-beta * (tt - tj));   // arg in [-beta, 0]: no edge cases
        av[e] = (_Float16)v;
    }
    v16h* dst = (v16h*)Tw + ((size_t)blk * 32 + l);
    *dst = av;
}

// ---------------------------------------------------------------------------
// Kernel 3: grid integral via WMMA. One wave (32 threads) per (batch, g-tile).
// S[ti, g] = sum_j T[ti,j] * P[g,j]  -> accumulate softplus(norm*S + mu)
// B fragments (32x16 f16): element e of lane half h holds K = (e<8?0:16) + h*8 + (e&7),
//                          the same order as the A fragments; n = lane&15.
// C layout: VGPR v -> M = v + 8*(lane>>4), N = lane&15.
// All 8 A fragments are preloaded before the WMMA chain so the b128 loads
// issue as one clause and their latency is covered by the B-side exp() work.
// 8*157 blocks x 32 threads
// ---------------------------------------------------------------------------
__global__ void k_grid(const float* __restrict__ x,
                       const float* __restrict__ pmu, const float* __restrict__ palpha,
                       const float* __restrict__ pbeta, const float* __restrict__ psigma,
                       const _Float16* __restrict__ Tw,
                       float* __restrict__ partial) {
    int blk = blockIdx.x;
    int gt = blk % GTILES;
    int b  = blk / GTILES;
    int l  = threadIdx.x;
    int h = l >> 4, lm = l & 15;

    __shared__ float sx[LQ], sy[LQ];
    __shared__ float red[32];
    const float* xb = x + (size_t)b * LQ * 3;
    for (int j = l; j < LQ; j += 32) {
        sx[j] = xb[j * 3 + 1];
        sy[j] = xb[j * 3 + 2];
    }

    // Preload all 8 A (temporal) fragments for this batch: 64 VGPRs.
    const v16h* Tv = (const v16h*)Tw;
    v16h A[8];
#pragma unroll
    for (int r = 0; r < 4; ++r) {
        A[2 * r + 0] = Tv[((size_t)(b * 4 + r) * 2 + 0) * 32 + l];
        A[2 * r + 1] = Tv[((size_t)(b * 4 + r) * 2 + 1) * 32 + l];
    }

    __syncthreads();

    float mu = *pmu, alpha = *palpha, beta = *pbeta, sigma = *psigma;
    float inv2s2 = 1.0f / (2.0f * sigma * sigma);
    float norm   = alpha * beta / (2.0f * PI_F * sigma * sigma);

    int g  = gt * 16 + lm;
    int gc = (g < NGP) ? g : (NGP - 1);
    float gx = (float)(gc / 50) * (1.0f / 49.0f);
    float gy = (float)(gc % 50) * (1.0f / 49.0f);
    float colmask = (g < NGP) ? 1.0f : 0.0f;

    // spatial factor fragments: P[g, j] for j = 0..63.
    // Values land in f16, so native-rate __expf is accuracy-neutral here.
    v16h Blo, Bhi;
#pragma unroll
    for (int e = 0; e < 16; ++e) {
        int jlo = ((e < 8) ? 0 : 16) + h * 8 + (e & 7);        // K = (e<8?0:16) + h*8 + (e&7)
        int jhi = 32 + ((e < 8) ? 0 : 16) + h * 8 + (e & 7);
        float dx = gx - sx[jlo], dy = gy - sy[jlo];
        Blo[e] = (_Float16)__expf(-(dx * dx + dy * dy) * inv2s2);
        dx = gx - sx[jhi]; dy = gy - sy[jhi];
        Bhi[e] = (_Float16)__expf(-(dx * dx + dy * dy) * inv2s2);
    }

    float local = 0.0f;
#pragma unroll
    for (int r = 0; r < 4; ++r) {
        v8f c = {};
        c = __builtin_amdgcn_wmma_f32_16x16x32_f16(false, A[2 * r + 0], false, Blo,
                                                   (short)0, c, false, false);
        c = __builtin_amdgcn_wmma_f32_16x16x32_f16(false, A[2 * r + 1], false, Bhi,
                                                   (short)0, c, false, false);
#pragma unroll
        for (int v = 0; v < 8; ++v) {
            // row index mg = r*16 + v + 8*h; constant-folds to "valid" for r<3,
            // and to a simple h/v compare for r==3. Branchless multiply mask:
            // padded rows carry T==0 so softplus(mu) is finite garbage we drop.
            float rowmask = ((r * 16 + v + 8 * h) < NTI) ? 1.0f : 0.0f;
            local += (rowmask * colmask) * softplus_f(norm * c[v] + mu);
        }
    }

    red[l] = local;
    __syncthreads();
    for (int s = 16; s > 0; s >>= 1) {
        if (l < s) red[l] += red[l + s];
        __syncthreads();
    }
    if (l == 0) partial[blk] = red[0];   // deterministic: no float atomics
}

// ---------------------------------------------------------------------------
// Kernel 4: loglik[b] = sumlog[b] - (sum of partials) * UNIT_VOL
// 8 blocks x 256 threads
// ---------------------------------------------------------------------------
__global__ void k_final(const float* __restrict__ sumlog,
                        const float* __restrict__ partial,
                        float* __restrict__ loglik) {
    int b = blockIdx.x;
    int tix = threadIdx.x;
    __shared__ float red[256];
    float a = 0.0f;
    for (int i = tix; i < GTILES; i += 256) a += partial[(size_t)b * GTILES + i];
    red[tix] = a;
    __syncthreads();
    for (int s = 128; s > 0; s >>= 1) {
        if (tix < s) red[tix] += red[tix + s];
        __syncthreads();
    }
    if (tix == 0) loglik[b] = sumlog[b] - red[0] * UNIT_VOL_F;
}

// ---------------------------------------------------------------------------
extern "C" void kernel_launch(void* const* d_in, const int* in_sizes, int n_in,
                              void* d_out, int out_size, void* d_ws, size_t ws_size,
                              hipStream_t stream) {
    const float* x      = (const float*)d_in[0];
    const float* pmu    = (const float*)d_in[1];
    const float* palpha = (const float*)d_in[2];
    const float* pbeta  = (const float*)d_in[3];
    const float* psigma = (const float*)d_in[4];

    float* out = (float*)d_out;                 // [0,512): lams ; [512,520): loglik

    // workspace layout
    char* ws = (char*)d_ws;
    _Float16* Tw   = (_Float16*)ws;                                   // 8*4*2*32*16 halves = 64 KB
    float* sumlog  = (float*)(ws + (size_t)BQ * 4 * 2 * 32 * 16 * sizeof(_Float16));
    float* partial = sumlog + BQ;                                     // 8*157 floats

    k_lams <<<BQ, LQ, 0, stream>>>(x, pmu, palpha, pbeta, psigma, out, sumlog);
    k_tfrag<<<BQ * 4 * 2, 32, 0, stream>>>(x, pbeta, Tw);
    k_grid <<<BQ * GTILES, 32, 0, stream>>>(x, pmu, palpha, pbeta, psigma, Tw, partial);
    k_final<<<BQ, 256, 0, stream>>>(sumlog, partial, out + BQ * LQ);
}
